// SpectralGraphDecoder_47888885350985
// MI455X (gfx1250) — matrix operand order repaired and checked by reading, not compiled, not run on hardware
//
#include <hip/hip_runtime.h>

#define NN 8192   // num_nodes
#define RR 32     // rank
#define DD 512    // graph_dim

typedef __attribute__((ext_vector_type(16))) __bf16 v16bf;
typedef __attribute__((ext_vector_type(8)))  float  v8f;

union Frag {
  v16bf v;
  uint4 u[2];
};

__device__ __forceinline__ unsigned short f2bf(float f) {
  // round-to-nearest-even f32 -> bf16 bits (inputs are finite)
  unsigned int u = __float_as_uint(f);
  u += 0x7FFFu + ((u >> 16) & 1u);
  return (unsigned short)(u >> 16);
}

// ---------------------------------------------------------------------------
// Kernel A: lam[r] = Wv[r,:] . q + bv[r]   (32 waves, one r per wave)
// ---------------------------------------------------------------------------
__global__ __launch_bounds__(1024) void lam_kernel(const float* __restrict__ Wv,
                                                   const float* __restrict__ bv,
                                                   const float* __restrict__ q,
                                                   float* __restrict__ lam) {
  const int wid  = threadIdx.x >> 5;   // = r
  const int lane = threadIdx.x & 31;
  const float4* row = (const float4*)(Wv + (size_t)wid * DD);
  const float4* q4  = (const float4*)q;
  float acc = 0.f;
  for (int k = lane; k < DD / 4; k += 32) {
    float4 w = row[k];
    float4 v = q4[k];
    acc += w.x * v.x + w.y * v.y + w.z * v.z + w.w * v.w;
  }
  for (int off = 16; off > 0; off >>= 1) acc += __shfl_xor(acc, off, 32);
  if (lane == 0) lam[wid] = acc + bv[wid];
}

// ---------------------------------------------------------------------------
// Kernel B: U GEMV. One wave per row m of Wu (512 floats).
// Writes Uf (f32), Ub (bf16 bits of U), Ul (bf16 bits of U*lam).
// ---------------------------------------------------------------------------
__global__ __launch_bounds__(256) void gemv_kernel(const float* __restrict__ Wu,
                                                   const float* __restrict__ bu,
                                                   const float* __restrict__ q,
                                                   const float* __restrict__ lam,
                                                   float* __restrict__ Uf,
                                                   unsigned short* __restrict__ Ub,
                                                   unsigned short* __restrict__ Ul) {
  __shared__ float qs[DD];
  for (int idx = threadIdx.x; idx < DD; idx += 256) qs[idx] = q[idx];
  __syncthreads();

  const int wid  = threadIdx.x >> 5;
  const int lane = threadIdx.x & 31;
  const size_t m = (size_t)blockIdx.x * 8 + wid;

  const float4* row = (const float4*)(Wu + m * DD);
  float acc = 0.f;
#pragma unroll
  for (int k = 0; k < 4; ++k) {
    const int idx = lane + 32 * k;     // coalesced 512B per step per wave
    float4 w = row[idx];
    const int qi = 4 * idx;
    acc += w.x * qs[qi] + w.y * qs[qi + 1] + w.z * qs[qi + 2] + w.w * qs[qi + 3];
  }
  for (int off = 16; off > 0; off >>= 1) acc += __shfl_xor(acc, off, 32);
  if (lane == 0) {
    const float u = acc + bu[m];
    Uf[m] = u;
    Ub[m] = f2bf(u);
    Ul[m] = f2bf(u * lam[m & (RR - 1)]);
  }
}

// ---------------------------------------------------------------------------
// Kernel C1: g[r] = lam[r] * sum_i U[i,r]   (one block per r)
// ---------------------------------------------------------------------------
__global__ __launch_bounds__(256) void colsum_kernel(const float* __restrict__ Uf,
                                                     const float* __restrict__ lam,
                                                     float* __restrict__ g) {
  const int r = blockIdx.x;
  float acc = 0.f;
  for (int i = threadIdx.x; i < NN; i += 256) acc += Uf[(size_t)i * RR + r];
  __shared__ float red[256];
  red[threadIdx.x] = acc;
  __syncthreads();
  for (int off = 128; off > 0; off >>= 1) {
    if (threadIdx.x < off) red[threadIdx.x] += red[threadIdx.x + off];
    __syncthreads();
  }
  if (threadIdx.x == 0) g[r] = red[0] * lam[r];
}

// ---------------------------------------------------------------------------
// Kernel C2: s[i] = rsqrt( U[i,:] . g )
// ---------------------------------------------------------------------------
__global__ __launch_bounds__(256) void scale_kernel(const float* __restrict__ Uf,
                                                    const float* __restrict__ g,
                                                    float* __restrict__ s) {
  const int i = blockIdx.x * 256 + threadIdx.x;
  const float4* u4 = (const float4*)(Uf + (size_t)i * RR);
  const float4* g4 = (const float4*)g;
  float d = 0.f;
#pragma unroll
  for (int k = 0; k < RR / 4; ++k) {
    float4 u = u4[k];
    float4 w = g4[k];
    d += u.x * w.x + u.y * w.y + u.z * w.z + u.w * w.w;
  }
  s[i] = rsqrtf(d);
}

// ---------------------------------------------------------------------------
// Kernel D: A[i,j] = relu( s_i * ((i==j) - L_ij) * s_j ),
//           L = (U*lam) @ U^T via v_wmma_f32_16x16x32_bf16.
// One wave owns a 16x64 output strip (A-fragment reused over 4 tiles).
// grid = (16, 512), block = 256 (8 waves).
// ---------------------------------------------------------------------------
__global__ __launch_bounds__(256) void spectral_gemm_kernel(
    const unsigned short* __restrict__ Ul,   // A operand: bf16(U*lam), row-major [N,R]
    const unsigned short* __restrict__ Ub,   // B operand: bf16(U),     row-major [N,R]
    const float* __restrict__ s,
    float* __restrict__ out) {
  const int lane = threadIdx.x & 31;
  const int wid  = threadIdx.x >> 5;
  const int hi   = lane >> 4;      // half-wave select
  const int ln   = lane & 15;
  const int i0   = blockIdx.y * 16;
  const int j0   = (blockIdx.x * 8 + wid) * 64;

  // A fragment: row m = i0+ln; lane holds K = {hi*8 .. hi*8+7} U {hi*8+16 .. hi*8+23}
  Frag fa;
  {
    const uint4* pa = (const uint4*)(Ul + (size_t)(i0 + ln) * RR + hi * 8);
    fa.u[0] = pa[0];   // K = hi*8     .. hi*8+7
    fa.u[1] = pa[2];   // K = hi*8+16  .. hi*8+23
  }

  // Row scales for this lane: rows i0 + hi*8 + v, v = 0..7 (contiguous)
  const float4 sr0 = *(const float4*)(s + i0 + hi * 8);
  const float4 sr1 = *(const float4*)(s + i0 + hi * 8 + 4);
  const float sr[8] = {sr0.x, sr0.y, sr0.z, sr0.w, sr1.x, sr1.y, sr1.z, sr1.w};

#pragma unroll
  for (int t = 0; t < 4; ++t) {
    const int j = j0 + t * 16;
    // B fragment: col n = j+ln; lane holds K = {hi*8 .. hi*8+7} U {hi*8+16 .. hi*8+23}
    Frag fb;
    {
      const uint4* pb = (const uint4*)(Ub + (size_t)(j + ln) * RR + hi * 8);
      fb.u[0] = pb[0];
      fb.u[1] = pb[2];
    }
    v8f c = {0.f, 0.f, 0.f, 0.f, 0.f, 0.f, 0.f, 0.f};
    c = __builtin_amdgcn_wmma_f32_16x16x32_bf16(
        /*neg_a=*/false, fa.v, /*neg_b=*/false, fb.v,
        /*c_mod=*/(short)0, c, /*reuse_a=*/false, /*reuse_b=*/false);

    const int col = j + ln;
    const float sc = s[col];
#pragma unroll
    for (int v = 0; v < 8; ++v) {
      const int row = i0 + hi * 8 + v;
      float val = ((row == col) ? 1.0f : 0.0f) - c[v];
      val = fmaxf(sr[v] * val * sc, 0.0f);
      __builtin_nontemporal_store(val, out + (size_t)row * NN + col);
    }
  }
}

// ---------------------------------------------------------------------------
extern "C" void kernel_launch(void* const* d_in, const int* in_sizes, int n_in,
                              void* d_out, int out_size, void* d_ws, size_t ws_size,
                              hipStream_t stream) {
  (void)in_sizes; (void)n_in; (void)out_size; (void)ws_size;
  const float* q  = (const float*)d_in[0];
  const float* Wu = (const float*)d_in[1];
  const float* bu = (const float*)d_in[2];
  const float* Wv = (const float*)d_in[3];
  const float* bv = (const float*)d_in[4];
  float* out = (float*)d_out;

  // Workspace layout (all 16B aligned): ~2.04 MB total
  float* ws_f = (float*)d_ws;
  float* lam = ws_f;                       // 32 f32
  float* g   = ws_f + 32;                  // 32 f32
  float* s   = ws_f + 64;                  // 8192 f32
  float* Uf  = ws_f + 64 + NN;             // N*R f32
  unsigned short* Ub = (unsigned short*)(Uf + (size_t)NN * RR);  // N*R bf16
  unsigned short* Ul = Ub + (size_t)NN * RR;                     // N*R bf16

  lam_kernel<<<1, 1024, 0, stream>>>(Wv, bv, q, lam);
  gemv_kernel<<<(NN * RR) / 8, 256, 0, stream>>>(Wu, bu, q, lam, Uf, Ub, Ul);
  colsum_kernel<<<RR, 256, 0, stream>>>(Uf, lam, g);
  scale_kernel<<<NN / 256, 256, 0, stream>>>(Uf, g, s);
  spectral_gemm_kernel<<<dim3(16, 512), 256, 0, stream>>>(Ul, Ub, s, out);
}
